// SupervisedContrastiveLoss_25116968747115
// MI455X (gfx1250) — matrix operand order repaired and checked by reading, not compiled, not run on hardware
//
#include <hip/hip_runtime.h>
#include <hip/hip_bf16.h>

typedef __attribute__((ext_vector_type(16))) _Float16 v16h;
typedef __attribute__((ext_vector_type(8)))  _Float16 v8h;
typedef __attribute__((ext_vector_type(8)))  float    v8f;

#define N_ROWS 8192            // B*V
#define DIM    256             // D
#define K_STEPS 8              // DIM / 32
#define J_CHUNKS 8
#define TILES_PER_CHUNK (N_ROWS / 16 / J_CHUNKS)   // 64
#define SCALE_LOG2E_OVER_T 20.609929155556625f     // log2(e)/0.07
#define EPS 1e-8f

struct HalfPair { v8h lo, hi; };
union AB { v16h v; HalfPair h; };

// ---------------- zero scratch + output ----------------
__global__ void scl_zero_kernel(float* __restrict__ sum_all,
                                float* __restrict__ sum_pos,
                                float* __restrict__ out) {
  int i = blockIdx.x * blockDim.x + threadIdx.x;
  if (i < N_ROWS) { sum_all[i] = 0.0f; sum_pos[i] = 0.0f; }
  if (i == 0) out[0] = 0.0f;
}

// ---------------- fp32 L2-normalize -> f16 ----------------
__global__ __launch_bounds__(DIM) void scl_normalize_kernel(
    const float* __restrict__ feat, _Float16* __restrict__ out) {
  const int row = blockIdx.x;
  const int tid = threadIdx.x;
  float x = feat[(size_t)row * DIM + tid];
  float p = x * x;
  #pragma unroll
  for (int off = 16; off > 0; off >>= 1) p += __shfl_xor(p, off, 32);
  __shared__ float wsum[DIM / 32];
  const int wave = tid >> 5, lane = tid & 31;
  if (lane == 0) wsum[wave] = p;
  __syncthreads();
  float tot = 0.0f;
  #pragma unroll
  for (int w = 0; w < DIM / 32; ++w) tot += wsum[w];
  out[(size_t)row * DIM + tid] = (_Float16)(x * rsqrtf(tot));
}

// ---------------- fused WMMA gram + exp + masked row-sums ----------------
__global__ __launch_bounds__(256) void scl_simexp_wmma_kernel(
    const _Float16* __restrict__ X, const long long* __restrict__ labels,
    float* __restrict__ sum_all, float* __restrict__ sum_pos) {
  const int tid  = threadIdx.x;
  const int wave = tid >> 5;
  const int lane = tid & 31;
  const int hs   = lane >> 4;   // half-wave select
  const int lm   = lane & 15;

  const int iTile = blockIdx.x * 8 + wave;
  const int i0    = iTile * 16;

  // Preload all A fragments for this i-tile (ISA 16-bit A 16x32 layout:
  // lanes 0-15: K {0..7}U{16..23}; lanes 16-31: K {8..15}U{24..31})
  AB a[K_STEPS];
  {
    const _Float16* rowp = X + (size_t)(i0 + lm) * DIM;
    #pragma unroll
    for (int k = 0; k < K_STEPS; ++k) {
      const int base = k * 32 + hs * 8;
      a[k].h.lo = *(const v8h*)(rowp + base);
      a[k].h.hi = *(const v8h*)(rowp + base + 16);
    }
  }

  // Row labels for this lane's 8 accumulator rows (C/D layout rows)
  int labr[8];
  #pragma unroll
  for (int m = 0; m < 8; ++m)
    labr[m] = (int)labels[(i0 + hs * 8 + m) >> 1];   // V=2 repeat

  float sall[8], spos[8];
  #pragma unroll
  for (int m = 0; m < 8; ++m) { sall[m] = 0.0f; spos[m] = 0.0f; }

  const int jtBeg = blockIdx.y * TILES_PER_CHUNK;
  const int jtEnd = jtBeg + TILES_PER_CHUNK;
  for (int jt = jtBeg; jt < jtEnd; ++jt) {
    const int j0   = jt * 16;
    const int colJ = j0 + lm;
    const int labj = (int)labels[colJ >> 1];
    const _Float16* colp = X + (size_t)colJ * DIM;

    v8f c = {};
    #pragma unroll
    for (int k = 0; k < K_STEPS; ++k) {
      // ISA 16-bit B 32x16 layout: lanes 0-15 hold K {0..7}U{16..23}, lanes 16-31 K {8..15}U{24..31}
      AB b;
      const int base = k * 32 + hs * 8;
      b.h.lo = *(const v8h*)(colp + base);
      b.h.hi = *(const v8h*)(colp + base + 16);
      c = __builtin_amdgcn_wmma_f32_16x16x32_f16(
              /*neg_a=*/false, a[k].v, /*neg_b=*/false, b.v,
              /*c_mod=*/(short)0, c, /*reuse_a=*/false, /*reuse_b=*/false);
    }

    // Branchless epilogue: |arg| <= 20.61 so raw v_exp_f32 is safe; diagonal
    // and label masks become v_cndmask selects (no exec-mask branches).
    #pragma unroll
    for (int m = 0; m < 8; ++m) {
      const float e    = __builtin_amdgcn_exp2f(c[m] * SCALE_LOG2E_OVER_T);
      const int   rowg = i0 + hs * 8 + m;
      const float ea   = (rowg != colJ) ? e : 0.0f;   // drop diagonal
      sall[m] += ea;
      spos[m] += (labr[m] == labj) ? ea : 0.0f;       // positive-pair mask
    }
  }

  // Reduce across the 16 lanes of each half-wave (xor masks stay in-half)
  #pragma unroll
  for (int m = 0; m < 8; ++m) {
    #pragma unroll
    for (int off = 8; off > 0; off >>= 1) {
      sall[m] += __shfl_xor(sall[m], off, 32);
      spos[m] += __shfl_xor(spos[m], off, 32);
    }
  }
  if (lm == 0) {
    #pragma unroll
    for (int m = 0; m < 8; ++m) {
      const int rowg = i0 + hs * 8 + m;
      atomicAdd(&sum_all[rowg], sall[m]);
      atomicAdd(&sum_pos[rowg], spos[m]);
    }
  }
}

// ---------------- per-row loss + mean ----------------
__global__ void scl_finalize_kernel(const float* __restrict__ sum_all,
                                    const float* __restrict__ sum_pos,
                                    float* __restrict__ out) {
  const int i = blockIdx.x * blockDim.x + threadIdx.x;
  if (i >= N_ROWS) return;
  const float num = sum_pos[i] + EPS;       // numerator (diag already excluded)
  const float den = sum_all[i] + EPS;       // denominator (diag already excluded)
  const float loss = -logf(num / den);
  atomicAdd(out, loss * (1.0f / (float)N_ROWS));
}

extern "C" void kernel_launch(void* const* d_in, const int* in_sizes, int n_in,
                              void* d_out, int out_size, void* d_ws, size_t ws_size,
                              hipStream_t stream) {
  const float*     feat   = (const float*)d_in[0];       // [4096, 2, 256] f32
  const long long* labels = (const long long*)d_in[1];   // [4096] i64
  float*           out    = (float*)d_out;               // scalar f32

  _Float16* Xh      = (_Float16*)d_ws;                                   // 4 MB
  float*    sum_all = (float*)((char*)d_ws + (size_t)N_ROWS * DIM * 2);  // 32 KB
  float*    sum_pos = sum_all + N_ROWS;                                  // 32 KB

  hipLaunchKernelGGL(scl_zero_kernel, dim3((N_ROWS + 255) / 256), dim3(256), 0,
                     stream, sum_all, sum_pos, out);
  hipLaunchKernelGGL(scl_normalize_kernel, dim3(N_ROWS), dim3(DIM), 0,
                     stream, feat, Xh);
  hipLaunchKernelGGL(scl_simexp_wmma_kernel,
                     dim3(N_ROWS / 16 / 8, J_CHUNKS), dim3(256), 0,
                     stream, Xh, labels, sum_all, sum_pos);
  hipLaunchKernelGGL(scl_finalize_kernel, dim3(N_ROWS / 256), dim3(256), 0,
                     stream, sum_all, sum_pos, out);
}
